// QuantumAttention_65481071405664
// MI455X (gfx1250) — matrix operand order repaired and checked by reading, not compiled, not run on hardware
//
#include <hip/hip_runtime.h>

// ---------------------------------------------------------------------------
// QuantumAttention for MI455X (gfx1250, wave32, WMMA 16x16x32 f16)
//   B=4, T=2048, C=1024, H=128, D=8
//   q = cos(x + sum(theta));  per token: S = q q^T / sqrt(8); P = softmax(S);
//   O = P q;  out = O @ W^T
// Tricks:
//  * K=8 / N=8 zero padding lives in LDS -> all fragments are plain
//    unconditional ds_load_b128 (no EXEC masks, no register zero-fill).
//  * |scores| <= 8/sqrt(8) = 2.83, so softmax needs no max subtraction, and
//    exp2 args are bounded -> raw v_exp_f32 (__builtin_amdgcn_exp2f) is exact
//    w.r.t. the guarded libm expansion.
//  * qtpad row 8 = ones -> the O = P*q WMMA's padded column 8 computes the
//    softmax row-sum for free; normalize only the final 16x8 tile.
// ---------------------------------------------------------------------------

typedef _Float16 h4  __attribute__((ext_vector_type(4)));
typedef _Float16 h8  __attribute__((ext_vector_type(8)));
typedef _Float16 h16 __attribute__((ext_vector_type(16)));
typedef float    f8v __attribute__((ext_vector_type(8)));

union HU { h16 v; h8 h[2]; };

__device__ __forceinline__ f8v wmma_f16(h16 a, h16 b, f8v c) {
  // (neg_a, A, neg_b, B, c_mod, C, reuse_a, reuse_b)
  return __builtin_amdgcn_wmma_f32_16x16x32_f16(false, a, false, b,
                                                (short)0, c, false, false);
}

#define C_DIM 1024
#define H_DIM 128
#define D_DIM 8
#define NTOK  8192  // B*T
#define KPAD  32    // K padded 8 -> 32 for 16x16x32 WMMA

// ---------------------------------------------------------------------------
// Stage 0: W (f32 [1024][1024], torch [out,in]) -> f16 workspace copy.
// Row n of W is exactly the WMMA B-matrix data for output channel n.
// ---------------------------------------------------------------------------
__global__ __launch_bounds__(256) void qa_convert_w(const float* __restrict__ w,
                                                    _Float16* __restrict__ wh) {
  int i = (blockIdx.x * 256 + threadIdx.x) * 4;
  float4 v = *(const float4*)(w + i);
  h4 o = { (_Float16)v.x, (_Float16)v.y, (_Float16)v.z, (_Float16)v.w };
  *(h4*)(wh + i) = o;
}

// ---------------------------------------------------------------------------
// Stage 1: per-token fused quantum-measure + head-vs-head attention.
// One block (8 waves) per token; wave w owns S/O rows [16w, 16w+16).
// ---------------------------------------------------------------------------
__global__ __launch_bounds__(256) void qa_stage1(const float* __restrict__ x,
                                                 const float* __restrict__ theta,
                                                 _Float16* __restrict__ outO) {
  __shared__ __align__(16) _Float16 qpad [H_DIM * KPAD];   // q[h][k], k>=8 -> 0
  __shared__ __align__(16) _Float16 qtpad[16 * H_DIM];     // d<8: q^T; d==8: 1; d>8: 0
  __shared__ __align__(16) _Float16 ps   [H_DIM * H_DIM];  // unnormalized P

  const int token = blockIdx.x;
  const int tid   = threadIdx.x;
  const int lane  = tid & 31;
  const int wid   = tid >> 5;
  const int half  = lane >> 4;   // which 16-lane half
  const int l16   = lane & 15;

  // ---- fill the pad regions (once per block, vector stores) ---------------
  const h8 z8 = {};
  const h8 one8 = { (_Float16)1.0f, (_Float16)1.0f, (_Float16)1.0f, (_Float16)1.0f,
                    (_Float16)1.0f, (_Float16)1.0f, (_Float16)1.0f, (_Float16)1.0f };
#pragma unroll
  for (int i = 0; i < 2; ++i) {            // qpad cols 8..31 of all 128 rows
    int id  = tid + i * 256;               // 0..511
    int row = id >> 2;
    int seg = id & 3;
    if (seg != 0) *(h8*)(&qpad[row * KPAD + seg * 8]) = z8;
  }
  if (tid < 128)                           // qtpad rows 8..15 (row 8 = ones)
    *(h8*)(&qtpad[8 * H_DIM + tid * 8]) = (tid < 16) ? one8 : z8;

  float tsum = 0.f;
#pragma unroll
  for (int j = 0; j < D_DIM; ++j) tsum += theta[j];

  // q = cos(x + sum(theta)); scatter into padded row-major and transposed
  const float* xrow = x + (size_t)token * C_DIM;
#pragma unroll
  for (int i = 0; i < 4; ++i) {
    int c = tid + i * 256;                 // c = h*8 + d
    _Float16 qh = (_Float16)__cosf(xrow[c] + tsum);
    qpad[(c >> 3) * KPAD + (c & 7)] = qh;
    qtpad[(c & 7) * H_DIM + (c >> 3)] = qh;
  }
  __syncthreads();

  // ---- Gram: S(16x128) = q_slab (16x8) * q^T, K zero-padded in LDS --------
  // A 16x32 layout: lane row = l16; V0-3 = K half*8.., V4-7 = K 16+half*8..
  HU a;
  {
    const int row = (wid * 16 + l16) * KPAD;
    a.h[0] = *(const h8*)(&qpad[row + half * 8]);
    a.h[1] = *(const h8*)(&qpad[row + 16 + half * 8]);
  }

  f8v s[8];
#pragma unroll
  for (int nt = 0; nt < 8; ++nt) {
    // B 32x16 layout: lane col = l16; V0-3 = K half*8.., V4-7 = K 16+half*8..
    const int brow = (nt * 16 + l16) * KPAD;
    HU b;
    b.h[0] = *(const h8*)(&qpad[brow + half * 8]);
    b.h[1] = *(const h8*)(&qpad[brow + 16 + half * 8]);
    f8v c = {};
    s[nt] = wmma_f16(a.v, b.v, c);
  }

  // ---- unnormalized softmax numerator (|arg| <= 4.1 -> bare v_exp_f32) ----
  // C-layout: row = r + 8*half, col = nt*16 + l16
  const float kScaleLog2e = 0.35355339059327373f * 1.4426950408889634f;
#pragma unroll
  for (int r = 0; r < 8; ++r) {
    const int grow = wid * 16 + r + half * 8;
#pragma unroll
    for (int nt = 0; nt < 8; ++nt)
      ps[grow * H_DIM + nt * 16 + l16] =
          (_Float16)__builtin_amdgcn_exp2f(s[nt][r] * kScaleLog2e);
  }

  // ---- O(16x9->16) = P~(16x128) * [q | 1] : col 8 = softmax row-sum -------
  f8v acc = {};
#pragma unroll
  for (int kt = 0; kt < 4; ++kt) {
    const int prow = wid * 16 + l16;
    HU pa;
    pa.h[0] = *(const h8*)(&ps[prow * H_DIM + kt * 32 + half * 8]);
    pa.h[1] = *(const h8*)(&ps[prow * H_DIM + kt * 32 + 16 + half * 8]);
    const int kb = kt * 32 + half * 8;
    HU bq;
    bq.h[0] = *(const h8*)(&qtpad[l16 * H_DIM + kb]);
    bq.h[1] = *(const h8*)(&qtpad[l16 * H_DIM + kb + 16]);
    acc = wmma_f16(pa.v, bq.v, acc);
  }

  // normalize by the free row-sum (lane l16==8 of each half) and store f16
  const int bsrc = (lane & 16) | 8;
  _Float16* orow = outO + (size_t)token * C_DIM;
#pragma unroll
  for (int r = 0; r < 8; ++r) {
    float rs  = __shfl(acc[r], bsrc, 32);
    float val = acc[r] * __builtin_amdgcn_rcpf(rs);
    if (l16 < D_DIM) {
      int h = wid * 16 + r + half * 8;
      orow[h * D_DIM + l16] = (_Float16)val;
    }
  }
}

// ---------------------------------------------------------------------------
// Stage 2: Y[8192][1024] = O[8192][1024] @ W^T   (f16 in, f32 accum/out)
// 128x128 block tile, 8 waves of 32x64, K stepped by 32 through LDS.
// ---------------------------------------------------------------------------
__global__ __launch_bounds__(256) void qa_gemm(const _Float16* __restrict__ A,
                                               const _Float16* __restrict__ Bw,
                                               float* __restrict__ Y) {
  constexpr int LDT = 40;  // 32 + 8 halfs pad: keeps b128 reads 16B-aligned
  __shared__ __align__(16) _Float16 At[128 * LDT];
  __shared__ __align__(16) _Float16 Bt[128 * LDT];

  const int tid  = threadIdx.x;
  const int lane = tid & 31;
  const int wid  = tid >> 5;
  const int half = lane >> 4;
  const int l16  = lane & 15;
  const int m0   = blockIdx.x * 128;
  const int n0   = blockIdx.y * 128;
  const int wm   = (wid & 3) * 32;
  const int wn   = (wid >> 2) * 64;

  f8v acc[2][4] = {};

  for (int k0 = 0; k0 < C_DIM; k0 += 32) {
#pragma unroll
    for (int i = 0; i < 2; ++i) {                 // 512 b128 loads / tile pair
      int idx = tid + i * 256;
      int row = idx >> 2;
      int seg = (idx & 3) * 8;
      *(h8*)(&At[row * LDT + seg]) =
          *(const h8*)(A  + (size_t)(m0 + row) * C_DIM + k0 + seg);
      *(h8*)(&Bt[row * LDT + seg]) =
          *(const h8*)(Bw + (size_t)(n0 + row) * C_DIM + k0 + seg);
      // pull next k-tile toward L2/WGP$ while this tile computes
      if (k0 + 32 < C_DIM) {
        __builtin_prefetch(A  + (size_t)(m0 + row) * C_DIM + k0 + 32 + seg, 0, 3);
        __builtin_prefetch(Bw + (size_t)(n0 + row) * C_DIM + k0 + 32 + seg, 0, 3);
      }
    }
    __syncthreads();

    HU a[2];
#pragma unroll
    for (int im = 0; im < 2; ++im) {
      int row = wm + im * 16 + l16;
      a[im].h[0] = *(const h8*)(&At[row * LDT + half * 8]);
      a[im].h[1] = *(const h8*)(&At[row * LDT + 16 + half * 8]);
    }
    HU b[4];
#pragma unroll
    for (int in = 0; in < 4; ++in) {
      int col = wn + in * 16 + l16;
      b[in].h[0] = *(const h8*)(&Bt[col * LDT + half * 8]);
      b[in].h[1] = *(const h8*)(&Bt[col * LDT + 16 + half * 8]);
    }
#pragma unroll
    for (int im = 0; im < 2; ++im)
#pragma unroll
      for (int in = 0; in < 4; ++in)
        acc[im][in] = wmma_f16(a[im].v, b[in].v, acc[im][in]);
    __syncthreads();
  }

#pragma unroll
  for (int im = 0; im < 2; ++im)
#pragma unroll
    for (int in = 0; in < 4; ++in) {
      int col = n0 + wn + in * 16 + l16;
#pragma unroll
      for (int r = 0; r < 8; ++r) {
        int row = m0 + wm + im * 16 + r + half * 8;
        Y[(size_t)row * C_DIM + col] = acc[im][in][r];
      }
    }
}

// ---------------------------------------------------------------------------
extern "C" void kernel_launch(void* const* d_in, const int* in_sizes, int n_in,
                              void* d_out, int out_size, void* d_ws, size_t ws_size,
                              hipStream_t stream) {
  (void)in_sizes; (void)n_in; (void)out_size; (void)ws_size;
  const float* x     = (const float*)d_in[0];   // [4,2048,1024]
  const float* theta = (const float*)d_in[1];   // [8]
  const float* wproj = (const float*)d_in[2];   // [1024,1024] (out,in)
  float* out = (float*)d_out;                   // [4,2048,1024] f32

  // workspace: O f16 (16 MB) then W f16 (2 MB); fully rewritten every call
  _Float16* wsO = (_Float16*)d_ws;
  _Float16* wsW = (_Float16*)((char*)d_ws + (size_t)NTOK * C_DIM * sizeof(_Float16));

  qa_convert_w<<<C_DIM * C_DIM / (256 * 4), 256, 0, stream>>>(wproj, wsW);
  qa_stage1   <<<NTOK, 256, 0, stream>>>(x, theta, wsO);
  qa_gemm     <<<dim3(NTOK / 128, C_DIM / 128), 256, 0, stream>>>(wsO, wsW, out);
}
